// EfficientAttention_34265249087758
// MI455X (gfx1250) — matrix operand order repaired and checked by reading, not compiled, not run on hardware
//
#include <hip/hip_runtime.h>
#include <hip/hip_bf16.h>

#define N_SPATIAL 4096
#define DIM 256
#define CDIM 64
#define NBATCH 4
#define ATT_SCALE 0.125f

typedef __attribute__((ext_vector_type(16))) __bf16 v16bf;
typedef __attribute__((ext_vector_type(8)))  __bf16 v8bf;
typedef __attribute__((ext_vector_type(8)))  float  v8f;

// ---- WMMA fragment index helpers (CDNA5 ISA 7.12.2, wave32) ----
// A (16x32, 16-bit): lane%16 = M; this code packs each lane's 16 elements contiguously (not the ISA table order):
//   K = base..base+15, base = chunk + g*16
// B (32x16, 16-bit): lane%16 = N; element e -> K = e + g*16 (contiguous order chosen here, same as A)
// C/D (16x16 f32): VGPR r -> row r + g*8; col = lane%16

__device__ __forceinline__ v16bf join8(v8bf lo, v8bf hi) {
    return __builtin_shufflevector(lo, hi, 0, 1, 2, 3, 4, 5, 6, 7,
                                   8, 9, 10, 11, 12, 13, 14, 15);
}

__device__ __forceinline__ v16bf loadAfrag(const __bf16* rowBase) {
    v8bf lo = *(const v8bf*)rowBase;
    v8bf hi = *(const v8bf*)(rowBase + 8);
    return join8(lo, hi);
}

__device__ __forceinline__ void wait_async0() {
#if __has_builtin(__builtin_amdgcn_s_wait_asynccnt)
    __builtin_amdgcn_s_wait_asynccnt(0);
#else
    asm volatile("s_wait_asynccnt 0" ::: "memory");
#endif
}

// async 16B global->LDS copy (GLOBAL_LOAD_ASYNC_TO_LDS_B128, ASYNCcnt-tracked)
__device__ __forceinline__ void async_copy16(const void* gsrc, void* ldst) {
    unsigned int       lds = (unsigned int)(uintptr_t)ldst; // low 32b of generic = LDS addr
    unsigned long long ga  = (unsigned long long)(uintptr_t)gsrc;
    asm volatile("global_load_async_to_lds_b128 %0, %1, off"
                 :: "v"(lds), "v"(ga) : "memory");
}

// =====================================================================
// Prep A: x [b][d][n] f32  ->  xT [b][n][d] bf16   (32x32 LDS transpose)
// =====================================================================
__global__ __launch_bounds__(256) void xpose_kernel(
    const float* __restrict__ x, __bf16* __restrict__ xT)
{
    __shared__ __bf16 tile[32][33];
    const int tid = threadIdx.x;
    const int tx = tid & 31, ty = tid >> 5;                  // 32 x 8
    const int id  = blockIdx.x;
    const int b   = id / ((DIM / 32) * (N_SPATIAL / 32));
    const int rem = id % ((DIM / 32) * (N_SPATIAL / 32));
    const int d0  = (rem / (N_SPATIAL / 32)) * 32;
    const int n0  = (rem % (N_SPATIAL / 32)) * 32;
    const float* xb = x + (size_t)b * DIM * N_SPATIAL;
#pragma unroll
    for (int i = 0; i < 4; ++i) {
        const int d = ty + i * 8;
        tile[d][tx] = (__bf16)xb[(size_t)(d0 + d) * N_SPATIAL + n0 + tx];
    }
    __syncthreads();
#pragma unroll
    for (int i = 0; i < 4; ++i) {
        const int n = ty + i * 8;
        xT[((size_t)b * N_SPATIAL + n0 + n) * DIM + d0 + tx] = tile[tx][n];
    }
}

// =====================================================================
// Prep B: weights f32 -> bf16 (same layouts)
// =====================================================================
__global__ __launch_bounds__(256) void wconv_kernel(
    const float* __restrict__ qkv_w, const float* __restrict__ proj_w,
    __bf16* __restrict__ qw, __bf16* __restrict__ pw)
{
    const int i = blockIdx.x * 256 + threadIdx.x;
    if (i < 3 * CDIM * DIM) qw[i] = (__bf16)qkv_w[i];
    if (i < DIM * CDIM)     pw[i] = (__bf16)proj_w[i];
}

// =====================================================================
// Kernel 1: qkv projection -> q [b][n][c], k [b][n][c], vT [b][c][n] (bf16)
// one wave computes a 16(n) x 16(o) tile, K = 256 over 8 WMMA steps
// =====================================================================
__global__ __launch_bounds__(256) void qkv_proj_kernel(
    const __bf16* __restrict__ xT, const __bf16* __restrict__ qw,
    const float* __restrict__ qkv_b,
    __bf16* __restrict__ q, __bf16* __restrict__ k, __bf16* __restrict__ vT)
{
    const int lane = threadIdx.x & 31;
    const int wid  = threadIdx.x >> 5;
    const int g    = lane >> 4;
    const int r16  = lane & 15;
    const int id   = blockIdx.x * 8 + wid;
    const int b    = id / (256 * 12);
    const int rem  = id % (256 * 12);
    const int nbase = (rem / 12) * 16;
    const int obase = (rem % 12) * 16;
    const __bf16* xb = xT + (size_t)b * N_SPATIAL * DIM;

    v8f acc = {};
#pragma unroll
    for (int k0 = 0; k0 < DIM; k0 += 32) {
        // A[m][kk] = xT[b][nbase+m][kk]  (m = r16) — two b128 loads
        v16bf a  = loadAfrag(xb + (size_t)(nbase + r16) * DIM + k0 + g * 16);
        // B[kk][n] = qw[obase+n][kk]     (n = r16) — 16 contiguous bf16
        v16bf bb = *(const v16bf*)(qw + (size_t)(obase + r16) * DIM + k0 + g * 16);
        acc = __builtin_amdgcn_wmma_f32_16x16x32_bf16(false, a, false, bb,
                                                      (short)0, acc, false, false);
    }
    const float bias = qkv_b[obase + r16];
    const int comp = obase >> 6;            // wave-uniform: which of q/k/v
    const int oc   = (obase & 63) + r16;    // column inside that component
    if (comp == 0) {
#pragma unroll
        for (int rr = 0; rr < 8; ++rr)
            q[((size_t)b * N_SPATIAL + nbase + rr + (g ? 8 : 0)) * CDIM + oc] =
                (__bf16)(acc[rr] + bias);
    } else if (comp == 1) {
#pragma unroll
        for (int rr = 0; rr < 8; ++rr)
            k[((size_t)b * N_SPATIAL + nbase + rr + (g ? 8 : 0)) * CDIM + oc] =
                (__bf16)(acc[rr] + bias);
    } else {
#pragma unroll
        for (int rr = 0; rr < 8; ++rr)
            vT[((size_t)b * CDIM + oc) * N_SPATIAL + nbase + rr + (g ? 8 : 0)] =
                (__bf16)(acc[rr] + bias);
    }
}

// =====================================================================
// Kernel 2: flash attention. 16 queries/wave, 128/block, 32-key blocks
// staged into LDS with GLOBAL_LOAD_ASYNC_TO_LDS_B128. d = 64.
// =====================================================================
__global__ __launch_bounds__(256) void flash_attn_kernel(
    const __bf16* __restrict__ q, const __bf16* __restrict__ k,
    const __bf16* __restrict__ vT, __bf16* __restrict__ o_mid)
{
    __shared__ __align__(128) __bf16 kTile[32][CDIM];   // [key][d]   4 KB
    __shared__ __align__(128) __bf16 vTile[CDIM][32];   // [d][key]   4 KB
    __shared__ __align__(128) __bf16 pTile[8][16][32];  // per-wave P slab 8 KB

    const int tid  = threadIdx.x;
    const int lane = tid & 31;
    const int wid  = tid >> 5;
    const int g    = lane >> 4;
    const int r16  = lane & 15;
    const int b     = blockIdx.x >> 5;
    const int qbase = (blockIdx.x & 31) * 128 + wid * 16;

    const __bf16* qb = q  + (size_t)b * N_SPATIAL * CDIM;
    const __bf16* kb = k  + (size_t)b * N_SPATIAL * CDIM;
    const __bf16* vb = vT + (size_t)b * CDIM * N_SPATIAL;

    // Q as two A-fragments (d chunks [0,32) and [32,64))
    v16bf aQ[2];
#pragma unroll
    for (int ch = 0; ch < 2; ++ch)
        aQ[ch] = loadAfrag(qb + (size_t)(qbase + r16) * CDIM + ch * 32 + g * 16);

    float m_run[8], l_run[8];
    v8f acc[4];
#pragma unroll
    for (int rr = 0; rr < 8; ++rr) { m_run[rr] = -1e30f; l_run[rr] = 0.f; }
#pragma unroll
    for (int ch = 0; ch < 4; ++ch)
#pragma unroll
        for (int rr = 0; rr < 8; ++rr) acc[ch][rr] = 0.f;

    // per-thread staging assignments (16 B each, 4 KB per tile, 256 threads)
    const int vrow = tid >> 2, vchk = (tid & 3) * 8;

    for (int kb0 = 0; kb0 < N_SPATIAL; kb0 += 32) {
        __syncthreads();   // previous iteration's tile reads complete
        async_copy16(kb + (size_t)kb0 * CDIM + tid * 8, &kTile[0][0] + tid * 8);
        async_copy16(vb + (size_t)vrow * N_SPATIAL + kb0 + vchk, &vTile[vrow][vchk]);
        wait_async0();
        __syncthreads();

        // S = Q K^T for two 16-key tiles
        v8f S[2];
#pragma unroll
        for (int t = 0; t < 2; ++t) {
            v8f c = {};
#pragma unroll
            for (int ch = 0; ch < 2; ++ch) {
                v16bf bK = *(const v16bf*)(&kTile[t * 16 + r16][ch * 32 + g * 16]);
                c = __builtin_amdgcn_wmma_f32_16x16x32_bf16(false, aQ[ch], false, bK,
                                                            (short)0, c, false, false);
            }
            S[t] = c;
        }

        // online softmax: rows live across 16-lane halves -> shfl_xor width 16
        float alpha[8];
#pragma unroll
        for (int rr = 0; rr < 8; ++rr) {
            const float s0 = S[0][rr] * ATT_SCALE;
            const float s1 = S[1][rr] * ATT_SCALE;
            float mx = fmaxf(s0, s1);
#pragma unroll
            for (int off = 1; off < 16; off <<= 1)
                mx = fmaxf(mx, __shfl_xor(mx, off, 16));
            const float mn = fmaxf(m_run[rr], mx);
            alpha[rr] = __expf(m_run[rr] - mn);
            m_run[rr] = mn;
            const float p0 = __expf(s0 - mn);
            const float p1 = __expf(s1 - mn);
            S[0][rr] = p0; S[1][rr] = p1;
            float rs = p0 + p1;
#pragma unroll
            for (int off = 1; off < 16; off <<= 1)
                rs += __shfl_xor(rs, off, 16);
            l_run[rr] = l_run[rr] * alpha[rr] + rs;
        }
#pragma unroll
        for (int ch = 0; ch < 4; ++ch)
#pragma unroll
            for (int rr = 0; rr < 8; ++rr) acc[ch][rr] *= alpha[rr];

        // C-layout -> A-layout shuffle of P via per-wave LDS slab
#pragma unroll
        for (int rr = 0; rr < 8; ++rr) {
            pTile[wid][rr + (g ? 8 : 0)][r16]      = (__bf16)S[0][rr];
            pTile[wid][rr + (g ? 8 : 0)][16 + r16] = (__bf16)S[1][rr];
        }
        v16bf aP = loadAfrag(&pTile[wid][r16][g * 16]);

        // O += P * V  (four 16-wide d chunks)
#pragma unroll
        for (int ch = 0; ch < 4; ++ch) {
            v16bf bV = *(const v16bf*)(&vTile[ch * 16 + r16][g * 16]);
            acc[ch] = __builtin_amdgcn_wmma_f32_16x16x32_bf16(false, aP, false, bV,
                                                              (short)0, acc[ch], false, false);
        }
    }

    // normalize, store O as [b][n][c] bf16
#pragma unroll
    for (int ch = 0; ch < 4; ++ch)
#pragma unroll
        for (int rr = 0; rr < 8; ++rr) {
            const float val = acc[ch][rr] / l_run[rr];
            const int   nsp = qbase + rr + (g ? 8 : 0);
            o_mid[((size_t)b * N_SPATIAL + nsp) * CDIM + ch * 16 + r16] = (__bf16)val;
        }
}

// =====================================================================
// Kernel 3: out[b][o][n] = sum_c proj_w[o][c] * o_mid[b][n][c] + proj_b[o]
// =====================================================================
__global__ __launch_bounds__(256) void proj_kernel(
    const __bf16* __restrict__ o_mid, const __bf16* __restrict__ pw,
    const float* __restrict__ proj_b, float* __restrict__ out)
{
    const int lane = threadIdx.x & 31;
    const int wid  = threadIdx.x >> 5;
    const int g    = lane >> 4;
    const int r16  = lane & 15;
    const int id   = blockIdx.x * 8 + wid;
    const int b    = id / (256 * 16);
    const int rem  = id % (256 * 16);
    const int nbase = (rem >> 4) * 16;
    const int obase = (rem & 15) * 16;
    const __bf16* ob = o_mid + (size_t)b * N_SPATIAL * CDIM;

    v8f acc = {};
#pragma unroll
    for (int ch = 0; ch < 2; ++ch) {
        // A[m][c] = pw[obase+m][c] — two b128 loads
        v16bf a  = loadAfrag(pw + (size_t)(obase + r16) * CDIM + ch * 32 + g * 16);
        // B[c][n] = o_mid[b][nbase+n][c] — 16 contiguous bf16
        v16bf bb = *(const v16bf*)(ob + (size_t)(nbase + r16) * CDIM + ch * 32 + g * 16);
        acc = __builtin_amdgcn_wmma_f32_16x16x32_bf16(false, a, false, bb,
                                                      (short)0, acc, false, false);
    }
#pragma unroll
    for (int rr = 0; rr < 8; ++rr) {
        const int oo = obase + rr + (g ? 8 : 0);
        out[((size_t)b * DIM + oo) * N_SPATIAL + nbase + r16] = acc[rr] + proj_b[oo];
    }
}

// =====================================================================
extern "C" void kernel_launch(void* const* d_in, const int* in_sizes, int n_in,
                              void* d_out, int out_size, void* d_ws, size_t ws_size,
                              hipStream_t stream) {
    (void)in_sizes; (void)n_in; (void)out_size; (void)ws_size;
    const float* x      = (const float*)d_in[0];
    const float* qkv_w  = (const float*)d_in[1];
    const float* qkv_b  = (const float*)d_in[2];
    const float* proj_w = (const float*)d_in[3];
    const float* proj_b = (const float*)d_in[4];
    float* out = (float*)d_out;

    const size_t perArr = (size_t)NBATCH * N_SPATIAL * CDIM;   // elements (bf16)
    __bf16* q     = (__bf16*)d_ws;
    __bf16* k     = q  + perArr;
    __bf16* vT    = k  + perArr;
    __bf16* o_mid = vT + perArr;
    __bf16* xT    = o_mid + perArr;                            // [b][n][d]  8 MB
    __bf16* qw    = xT + (size_t)NBATCH * N_SPATIAL * DIM;     // 192x256
    __bf16* pw    = qw + (size_t)3 * CDIM * DIM;               // 256x64

    // prep: transpose/convert x, convert weights
    xpose_kernel<<<NBATCH * (DIM / 32) * (N_SPATIAL / 32), 256, 0, stream>>>(x, xT);
    wconv_kernel<<<(3 * CDIM * DIM + 255) / 256, 256, 0, stream>>>(qkv_w, proj_w, qw, pw);

    qkv_proj_kernel<<<12288 / 8, 256, 0, stream>>>(xT, qw, qkv_b, q, k, vT);
    flash_attn_kernel<<<NBATCH * 32, 256, 0, stream>>>(q, k, vT, o_mid);
    proj_kernel<<<16384 / 8, 256, 0, stream>>>(o_mid, pw, proj_b, out);
}
